// GraphiT_Perceiver_Layer_72404558676376
// MI455X (gfx1250) — matrix operand order repaired and checked by reading, not compiled, not run on hardware
//
// GraphiT + Perceiver layer for MI455X (gfx1250, wave32, WMMA bf16).
// Key optimization: E = e@we (134MB) is never materialized; scores are computed
// as (e_slab @ (we ⊙ Q_row)) folded with K, so `e` is streamed exactly once.
// The e-stream uses CDNA5 async global->LDS DMA (double-buffered, ASYNCcnt).
#include <hip/hip_runtime.h>
#include <math.h>

typedef __attribute__((ext_vector_type(16))) __bf16 v16bf;
typedef __attribute__((ext_vector_type(8)))  float  v8f;

#define DI __device__ __forceinline__

#if defined(__has_builtin)
#if __has_builtin(__builtin_amdgcn_global_load_async_to_lds_b128)
#define HAVE_ASYNC_LDS 1
#endif
#endif

#ifndef HAVE_ASYNC_LDS
#define HAVE_ASYNC_LDS 0
#endif

#if defined(__has_builtin)
#if __has_builtin(__builtin_amdgcn_s_wait_asynccnt)
#define WAIT_ASYNC(n) __builtin_amdgcn_s_wait_asynccnt(n)
#endif
#endif
#ifndef WAIT_ASYNC
#define WAIT_ASYNC(n) asm volatile("s_wait_asynccnt %0" ::"n"(n) : "memory")
#endif

namespace {
constexpr int B_ = 8, N_ = 256, H_ = 8, DC_ = 64, NL_ = 32, LLAT_ = 2, HD_ = 64;
constexpr float SCALE_ = 0.35355339059327373f;  // DH^-0.5

// workspace layout (float offsets)
constexpr size_t OFF_QP    = 0;
constexpr size_t OFF_KP    = OFF_QP    + (size_t)B_ * N_ * HD_;
constexpr size_t OFF_VP    = OFF_KP    + (size_t)B_ * N_ * HD_;
constexpr size_t OFF_S     = OFF_VP    + (size_t)B_ * N_ * HD_;
constexpr size_t OFF_GMAX  = OFF_S     + (size_t)B_ * H_ * N_ * N_;
constexpr size_t OFF_PATTN = OFF_GMAX  + 64;
constexpr size_t OFF_HC    = OFF_PATTN + (size_t)B_ * N_ * HD_;
constexpr size_t OFF_KV    = OFF_HC    + (size_t)B_ * N_ * HD_;
constexpr size_t OFF_QBUF  = OFF_KV    + (size_t)B_ * N_ * 2 * HD_;

// monotone float<->uint key for atomicMax on floats
DI unsigned fkey(float f) {
  unsigned u = __float_as_uint(f);
  return (u & 0x80000000u) ? ~u : (u | 0x80000000u);
}
DI float fdec(unsigned k) {
  unsigned u = (k & 0x80000000u) ? (k & 0x7fffffffu) : ~k;
  return __uint_as_float(u);
}

DI v8f wmma_bf16(v16bf a, v16bf b, v8f c) {
  return __builtin_amdgcn_wmma_f32_16x16x32_bf16(false, a, false, b, (short)0, c,
                                                 false, false);
}

// A-matrix 16x32 bf16 fragment (ISA 7.12.2): lane m=lane&15, VGPR r holds
// k-pairs {2r,2r+1} (+16 for r>=4), +8 if lane>=16.
DI v16bf load_a_frag(const __bf16* base, int ld, int lane, int kk) {
  int m = lane & 15, hi = lane >> 4;
  v16bf a;
#pragma unroll
  for (int r = 0; r < 8; ++r) {
    int kb = ((r < 4) ? (2 * r) : (16 + 2 * (r - 4))) + hi * 8 + kk;
    a[2 * r]     = base[m * ld + kb];
    a[2 * r + 1] = base[m * ld + kb + 1];
  }
  return a;
}

// Same A layout but sourced from fp32 LDS (convert at fragment-build time).
DI v16bf load_a_frag_f32(const float* base, int ld, int lane, int kk) {
  int m = lane & 15, hi = lane >> 4;
  v16bf a;
#pragma unroll
  for (int r = 0; r < 8; ++r) {
    int kb = ((r < 4) ? (2 * r) : (16 + 2 * (r - 4))) + hi * 8 + kk;
    a[2 * r]     = (__bf16)base[m * ld + kb];
    a[2 * r + 1] = (__bf16)base[m * ld + kb + 1];
  }
  return a;
}

// B-matrix 32x16 bf16 fragment: lane n=lane&15; same k order as A fragment:
// VGPR v holds k-pairs {2v,2v+1} (+16 for v>=4), +8 if lane>=16.
DI v16bf load_b_frag(const __bf16* base, int ld, int lane, int kk) {
  int n = lane & 15, hi = lane >> 4;
  v16bf b;
#pragma unroll
  for (int v = 0; v < 8; ++v) {
    int k0 = kk + ((v < 4) ? (2 * v) : (16 + 2 * (v - 4))) + hi * 8;
    b[2 * v]     = base[k0 * ld + n];
    b[2 * v + 1] = base[(k0 + 1) * ld + n];
  }
  return b;
}

#if HAVE_ASYNC_LDS
// The builtin wants pointers to 4xint vectors (per hipcc diagnostic):
// arg0 = global (__device__, AS1) v4i*, arg1 = LDS (__shared__, AS3) v4i*.
typedef int v4i_ __attribute__((vector_size(16)));
typedef __attribute__((address_space(1))) v4i_* g_v4i_ptr;
typedef __attribute__((address_space(3))) v4i_* l_v4i_ptr;
// 16B per lane, global -> LDS DMA, tracked by ASYNCcnt.
DI void async_copy16(const float* g, float* l) {
  __builtin_amdgcn_global_load_async_to_lds_b128((g_v4i_ptr)(v4i_*)(void*)g,
                                                 (l_v4i_ptr)(v4i_*)(void*)l, 0, 0);
}
#endif
}  // namespace

__global__ void init_gmax_kernel(unsigned* gmax) {
  if (threadIdx.x < 64) gmax[threadIdx.x] = 0u;  // key 0 == -inf
}

// C[M,Nc] = A[M,64] @ W[64,Nc]; 256 threads = 8 waves, each wave one 16x16 tile.
// epi: 0 plain, 1 scale by SCALE_, 2 out = aux1 + aux2 + tanh(acc) (Nc==64).
__global__ void gemm_k64_bf16(const float* __restrict__ A, const float* __restrict__ W,
                              float* __restrict__ C, int Nc, int epi,
                              const float* __restrict__ aux1,
                              const float* __restrict__ aux2) {
  __shared__ __bf16 Al[32][64];
  __shared__ __bf16 Wl[64][64];
  const int tid = threadIdx.x;
  const int rowbase = blockIdx.x * 32;
  const int coloff = blockIdx.y * 64;
  for (int idx = tid; idx < 32 * 64; idx += 256)
    Al[idx >> 6][idx & 63] = (__bf16)A[(size_t)(rowbase + (idx >> 6)) * 64 + (idx & 63)];
  for (int idx = tid; idx < 64 * 64; idx += 256)
    Wl[idx >> 6][idx & 63] = (__bf16)W[(size_t)(idx >> 6) * Nc + coloff + (idx & 63)];
  __syncthreads();

  const int wave = tid >> 5, lane = tid & 31;
  const int mtile = wave & 1, ntile = wave >> 1;
  v16bf a0 = load_a_frag(&Al[mtile * 16][0], 64, lane, 0);
  v16bf a1 = load_a_frag(&Al[mtile * 16][0], 64, lane, 32);
  v16bf b0 = load_b_frag(&Wl[0][ntile * 16], 64, lane, 0);
  v16bf b1 = load_b_frag(&Wl[0][ntile * 16], 64, lane, 32);
  v8f acc = {};
  acc = wmma_bf16(a0, b0, acc);
  acc = wmma_bf16(a1, b1, acc);

  const int n = lane & 15, hi = lane >> 4;
#pragma unroll
  for (int r = 0; r < 8; ++r) {
    int row = rowbase + mtile * 16 + r + 8 * hi;
    int col = coloff + ntile * 16 + n;
    float v = acc[r];
    if (epi == 1) v *= SCALE_;
    if (epi == 2)
      v = aux1[(size_t)row * 64 + col] + aux2[(size_t)row * 64 + col] + tanhf(v);
    C[(size_t)row * Nc + col] = v;
  }
}

// Per (b,i): F = e[b,i,:,:](256x64) @ U (U = we ⊙ Q_row), scores = <F_head, K_head>.
// e is streamed via async global->LDS DMA, double-buffered (2 x 128-row chunks).
__global__ void edge_scores_kernel(const float* __restrict__ e,
                                   const float* __restrict__ we,
                                   const float* __restrict__ Qp,
                                   const float* __restrict__ Kp,
                                   float* __restrict__ S,
                                   unsigned* __restrict__ gmax) {
  __shared__ float Eb[2][128][64];  // fp32 stage; F (fp32) written back in place
  __shared__ __bf16 U[64][64];
  __shared__ unsigned smax[8];
  const int tid = threadIdx.x;
  const int b = blockIdx.x >> 8;
  const int i = blockIdx.x & 255;
  const float* ebase = e + (size_t)(b * N_ + i) * N_ * 64;

#if HAVE_ASYNC_LDS
  // Kick off DMA for both 32KB chunks first; U-build overlaps the transfers.
  // Each wave issues 8 async instructions per chunk (32 lanes x 16B each).
#pragma unroll
  for (int q = tid; q < 2048; q += 256) async_copy16(ebase + q * 4, &Eb[0][0][0] + q * 4);
#pragma unroll
  for (int q = tid; q < 2048; q += 256)
    async_copy16(ebase + 8192 + q * 4, &Eb[1][0][0] + q * 4);
#endif

  const float* qrow = Qp + (size_t)(b * N_ + i) * HD_;
  for (int idx = tid; idx < 64 * 64; idx += 256) {
    int c = idx >> 6, n = idx & 63;
    U[c][n] = (__bf16)(we[(size_t)c * 64 + n] * qrow[n]);
  }
  if (tid < 8) smax[tid] = 0u;

#if !HAVE_ASYNC_LDS
  {  // fallback: synchronous cooperative staging of both chunks
    const float4* e4 = (const float4*)ebase;
    float4* l4 = (float4*)&Eb[0][0][0];
#pragma unroll
    for (int q = tid; q < 4096; q += 256) l4[q] = e4[q];
  }
#endif

  const int wave = tid >> 5, lane = tid & 31;
  float maxv[4] = {-3.4e38f, -3.4e38f, -3.4e38f, -3.4e38f};

  for (int chunk = 0; chunk < 2; ++chunk) {
#if HAVE_ASYNC_LDS
    if (chunk == 0) { WAIT_ASYNC(8); } else { WAIT_ASYNC(0); }
#endif
    __syncthreads();
    const int j0c = chunk * 128;
    const int jloc = wave * 16;  // each wave owns one 16-row tile of the chunk
    v16bf a0 = load_a_frag_f32(&Eb[chunk][jloc][0], 64, lane, 0);
    v16bf a1 = load_a_frag_f32(&Eb[chunk][jloc][0], 64, lane, 32);
    const int n = lane & 15, hi = lane >> 4;
#pragma unroll
    for (int t = 0; t < 4; ++t) {
      v16bf b0 = load_b_frag(&U[0][t * 16], 64, lane, 0);
      v16bf b1 = load_b_frag(&U[0][t * 16], 64, lane, 32);
      v8f c = {};
      c = wmma_bf16(a0, b0, c);
      c = wmma_bf16(a1, b1, c);
#pragma unroll
      for (int r = 0; r < 8; ++r)  // write F (fp32) back into own LDS slice
        Eb[chunk][jloc + r + 8 * hi][t * 16 + n] = c[r];
    }
    // fold with K per head: s[h,j] = sum_k F[j,8h+k]*Kp[j,8h+k]
#pragma unroll
    for (int iter = 0; iter < 4; ++iter) {
      int hh = (lane >> 4) + 2 * iter;
      int jl = lane & 15;
      int jg = j0c + jloc + jl;
      const float* kr = Kp + (size_t)(b * N_ + jg) * HD_ + hh * 8;
      float s = 0.f;
#pragma unroll
      for (int k = 0; k < 8; ++k) s += Eb[chunk][jloc + jl][hh * 8 + k] * kr[k];
      S[((size_t)(b * H_ + hh) * N_ + i) * N_ + jg] = s;
      maxv[iter] = fmaxf(maxv[iter], s);
    }
  }
  __syncthreads();
#pragma unroll
  for (int iter = 0; iter < 4; ++iter)
    atomicMax(&smax[(lane >> 4) + 2 * iter], fkey(maxv[iter]));
  __syncthreads();
  if (tid < 8) atomicMax(&gmax[b * 8 + tid], smax[tid]);
}

// One wave per (b,h,itile): P = exp(S-max)*mask_i*mask_j*k_RW; out = P@[V|1]/denom.
__global__ void p_attn_apply_kernel(const float* __restrict__ S,
                                    const unsigned* __restrict__ gmax,
                                    const float* __restrict__ kRW,
                                    const float* __restrict__ mask,
                                    const float* __restrict__ Vp,
                                    float* __restrict__ pattn) {
  const int bid = blockIdx.x;
  const int b = bid >> 7, hh = (bid >> 4) & 7, it = bid & 15;
  const int lane = threadIdx.x;
  const float gmx = fdec(gmax[b * 8 + hh]);
  const int m = lane & 15, hi = lane >> 4, n = lane & 15;
  const int irow = it * 16 + m;
  const float mi = mask[b * N_ + irow];
  const float* Srow = S + ((size_t)(b * H_ + hh) * N_ + irow) * N_;
  const float* krow = kRW + ((size_t)b * N_ + irow) * N_;
  v8f acc = {};
#pragma unroll
  for (int ch = 0; ch < 8; ++ch) {  // K = 256 over 8 x 32
    v16bf a;
#pragma unroll
    for (int r = 0; r < 8; ++r) {
      int j0 = ch * 32 + ((r < 4) ? (2 * r) : (16 + 2 * (r - 4))) + hi * 8;
      float w0 = expf(Srow[j0] - gmx) * mi * mask[b * N_ + j0] * krow[j0];
      float w1 = expf(Srow[j0 + 1] - gmx) * mi * mask[b * N_ + j0 + 1] * krow[j0 + 1];
      a[2 * r] = (__bf16)w0;
      a[2 * r + 1] = (__bf16)w1;
    }
    v16bf bf;
#pragma unroll
    for (int v = 0; v < 8; ++v) {
#pragma unroll
      for (int h2 = 0; h2 < 2; ++h2) {
        int j = ch * 32 + ((v < 4) ? (2 * v) : (16 + 2 * (v - 4))) + hi * 8 + h2;
        float val = (n < 8) ? Vp[(size_t)(b * N_ + j) * HD_ + hh * 8 + n]
                            : (n == 8 ? 1.0f : 0.0f);
        bf[2 * v + h2] = (__bf16)val;
      }
    }
    acc = wmma_bf16(a, bf, acc);
  }
#pragma unroll
  for (int r = 0; r < 8; ++r) {
    float cr = acc[r];
    float den = __shfl(cr, 8 | (lane & 16), 32);  // ones-column of same half
    den = fmaxf(den, 1e-6f);
    float val = cr / den;
    int row = it * 16 + r + 8 * hi;
    if (n < 8) pattn[(size_t)(b * N_ + row) * HD_ + hh * 8 + n] = val;
  }
}

// Perceiver cross-attention: latents(32) attend to nodes(256). One block per b.
__global__ void cross_attn_kernel(const float* __restrict__ queries,
                                  const float* __restrict__ cq_wq,
                                  const float* __restrict__ KV,
                                  const float* __restrict__ cq_wo,
                                  const float* __restrict__ cq_bo,
                                  float* __restrict__ qbuf) {
  __shared__ float Qc[NL_][HD_];
  __shared__ float AO[NL_][HD_];
  const int b = blockIdx.x, tid = threadIdx.x;
  const float* qin = queries + (size_t)b * NL_ * DC_;
  for (int idx = tid; idx < NL_ * HD_; idx += 256) {
    int i2 = idx >> 6, n = idx & 63;
    float s = 0.f;
    for (int k = 0; k < DC_; ++k) s += qin[i2 * DC_ + k] * cq_wq[(size_t)k * HD_ + n];
    Qc[i2][n] = s;
  }
  __syncthreads();
  {
    const int hh = tid >> 5, i2 = tid & 31;
    const float* kvb = KV + (size_t)b * N_ * 128;
    float mx = -3.4e38f;
    for (int j = 0; j < N_; ++j) {
      const float* kj = kvb + (size_t)j * 128 + hh * 8;
      float s = 0.f;
#pragma unroll
      for (int k = 0; k < 8; ++k) s += Qc[i2][hh * 8 + k] * kj[k];
      s = fminf(fmaxf(s * SCALE_, -5.f), 5.f);
      mx = fmaxf(mx, s);
    }
    float den = 0.f, acc[8] = {0, 0, 0, 0, 0, 0, 0, 0};
    for (int j = 0; j < N_; ++j) {
      const float* kj = kvb + (size_t)j * 128 + hh * 8;
      const float* vj = kvb + (size_t)j * 128 + 64 + hh * 8;
      float s = 0.f;
#pragma unroll
      for (int k = 0; k < 8; ++k) s += Qc[i2][hh * 8 + k] * kj[k];
      s = fminf(fmaxf(s * SCALE_, -5.f), 5.f);
      float w = expf(s - mx);
      den += w;
#pragma unroll
      for (int d = 0; d < 8; ++d) acc[d] += w * vj[d];
    }
#pragma unroll
    for (int d = 0; d < 8; ++d) AO[i2][hh * 8 + d] = acc[d] / den;
  }
  __syncthreads();
  for (int idx = tid; idx < NL_ * DC_; idx += 256) {
    int i2 = idx >> 6, c = idx & 63;
    float s = cq_bo[c];
    for (int k = 0; k < HD_; ++k) s += AO[i2][k] * cq_wo[(size_t)k * DC_ + c];
    qbuf[(size_t)b * NL_ * DC_ + idx] = qin[idx] + s;
  }
}

// L latent self-attn layers + O_queries/LN1 + FFN + LN2. One block per b.
__global__ void latent_block_kernel(
    const float* __restrict__ qbuf, const float* __restrict__ sa_wq,
    const float* __restrict__ sa_wkv, const float* __restrict__ sa_wo,
    const float* __restrict__ sa_bo, const float* __restrict__ oq_w,
    const float* __restrict__ ln1_g, const float* __restrict__ ln1_b,
    const float* __restrict__ ffn_w1, const float* __restrict__ ffn_w2,
    const float* __restrict__ ln2_g, const float* __restrict__ ln2_b,
    float* __restrict__ out) {
  __shared__ float qc[NL_][DC_];
  __shared__ float Qh[NL_][HD_];
  __shared__ float Kh[NL_][HD_];
  __shared__ float Vh[NL_][HD_];
  __shared__ float att[NL_][HD_];
  __shared__ float Hb[NL_][2 * DC_];
  const int b = blockIdx.x, tid = threadIdx.x;
  for (int idx = tid; idx < NL_ * DC_; idx += 256)
    qc[idx >> 6][idx & 63] = qbuf[(size_t)b * NL_ * DC_ + idx];
  __syncthreads();

  for (int l = 0; l < LLAT_; ++l) {
    const float* wq = sa_wq + (size_t)l * DC_ * HD_;
    const float* wkv = sa_wkv + (size_t)l * DC_ * 2 * HD_;
    const float* wo = sa_wo + (size_t)l * HD_ * DC_;
    const float* bo = sa_bo + (size_t)l * DC_;
    for (int idx = tid; idx < NL_ * HD_; idx += 256) {
      int i2 = idx >> 6, n = idx & 63;
      float sq = 0.f, sk = 0.f, sv = 0.f;
      for (int k = 0; k < DC_; ++k) {
        float x = qc[i2][k];
        sq += x * wq[(size_t)k * HD_ + n];
        sk += x * wkv[(size_t)k * 2 * HD_ + n];
        sv += x * wkv[(size_t)k * 2 * HD_ + HD_ + n];
      }
      Qh[i2][n] = sq; Kh[i2][n] = sk; Vh[i2][n] = sv;
    }
    __syncthreads();
    {
      const int hh = tid >> 5, i2 = tid & 31;
      float mx = -3.4e38f;
      for (int j = 0; j < NL_; ++j) {
        float s = 0.f;
#pragma unroll
        for (int k = 0; k < 8; ++k) s += Qh[i2][hh * 8 + k] * Kh[j][hh * 8 + k];
        s = fminf(fmaxf(s * SCALE_, -5.f), 5.f);
        mx = fmaxf(mx, s);
      }
      float den = 0.f, acc[8] = {0, 0, 0, 0, 0, 0, 0, 0};
      for (int j = 0; j < NL_; ++j) {
        float s = 0.f;
#pragma unroll
        for (int k = 0; k < 8; ++k) s += Qh[i2][hh * 8 + k] * Kh[j][hh * 8 + k];
        s = fminf(fmaxf(s * SCALE_, -5.f), 5.f);
        float w = expf(s - mx);
        den += w;
#pragma unroll
        for (int d = 0; d < 8; ++d) acc[d] += w * Vh[j][hh * 8 + d];
      }
#pragma unroll
      for (int d = 0; d < 8; ++d) att[i2][hh * 8 + d] = acc[d] / den;
    }
    __syncthreads();
    for (int idx = tid; idx < NL_ * DC_; idx += 256) {
      int i2 = idx >> 6, c = idx & 63;
      float s = bo[c];
      for (int k = 0; k < HD_; ++k) s += att[i2][k] * wo[(size_t)k * DC_ + c];
      qc[i2][c] = qc[i2][c] + s;  // residual; each thread RWs its own element
    }
    __syncthreads();
  }
  // q @ oq_w -> Qh (reuse)
  for (int idx = tid; idx < NL_ * DC_; idx += 256) {
    int i2 = idx >> 6, c = idx & 63;
    float s = 0.f;
    for (int k = 0; k < DC_; ++k) s += qc[i2][k] * oq_w[(size_t)k * DC_ + c];
    Qh[i2][c] = s;
  }
  __syncthreads();
  if (tid < NL_) {  // LN1 -> qc
    int i2 = tid;
    float m = 0.f;
    for (int c = 0; c < DC_; ++c) m += Qh[i2][c];
    m *= (1.0f / DC_);
    float v = 0.f;
    for (int c = 0; c < DC_; ++c) { float d = Qh[i2][c] - m; v += d * d; }
    v *= (1.0f / DC_);
    float rs = rsqrtf(v + 1e-5f);
    for (int c = 0; c < DC_; ++c)
      qc[i2][c] = (Qh[i2][c] - m) * rs * ln1_g[c] + ln1_b[c];
  }
  __syncthreads();
  for (int idx = tid; idx < NL_ * 2 * DC_; idx += 256) {  // FFN up + relu
    int i2 = idx >> 7, u = idx & 127;
    float s = 0.f;
    for (int k = 0; k < DC_; ++k) s += qc[i2][k] * ffn_w1[(size_t)k * 2 * DC_ + u];
    Hb[i2][u] = fmaxf(s, 0.f);
  }
  __syncthreads();
  for (int idx = tid; idx < NL_ * DC_; idx += 256) {  // FFN down + residual
    int i2 = idx >> 6, c = idx & 63;
    float s = 0.f;
    for (int k = 0; k < 2 * DC_; ++k) s += Hb[i2][k] * ffn_w2[(size_t)k * DC_ + c];
    att[i2][c] = qc[i2][c] + s;
  }
  __syncthreads();
  if (tid < NL_) {  // LN2 -> out
    int i2 = tid;
    float m = 0.f;
    for (int c = 0; c < DC_; ++c) m += att[i2][c];
    m *= (1.0f / DC_);
    float v = 0.f;
    for (int c = 0; c < DC_; ++c) { float d = att[i2][c] - m; v += d * d; }
    v *= (1.0f / DC_);
    float rs = rsqrtf(v + 1e-5f);
    for (int c = 0; c < DC_; ++c)
      out[((size_t)b * NL_ + i2) * DC_ + c] = (att[i2][c] - m) * rs * ln2_g[c] + ln2_b[c];
  }
}

extern "C" void kernel_launch(void* const* d_in, const int* in_sizes, int n_in,
                              void* d_out, int out_size, void* d_ws, size_t ws_size,
                              hipStream_t stream) {
  (void)in_sizes; (void)n_in; (void)out_size; (void)ws_size;
  const float* hfeat  = (const float*)d_in[0];
  const float* p      = (const float*)d_in[1];
  const float* e      = (const float*)d_in[2];
  const float* k_RW   = (const float*)d_in[3];
  const float* queries= (const float*)d_in[4];
  const float* mask   = (const float*)d_in[5];
  const float* wq_p   = (const float*)d_in[6];
  const float* wk_p   = (const float*)d_in[7];
  const float* we_p   = (const float*)d_in[8];
  const float* wv_p   = (const float*)d_in[9];
  const float* wo_p   = (const float*)d_in[10];
  const float* cq_wq  = (const float*)d_in[11];
  const float* cq_wkv = (const float*)d_in[12];
  const float* cq_wo  = (const float*)d_in[13];
  const float* cq_bo  = (const float*)d_in[14];
  const float* sa_wq  = (const float*)d_in[15];
  const float* sa_wkv = (const float*)d_in[16];
  const float* sa_wo  = (const float*)d_in[17];
  const float* sa_bo  = (const float*)d_in[18];
  const float* oq_w   = (const float*)d_in[19];
  const float* ln1_g  = (const float*)d_in[20];
  const float* ln1_b  = (const float*)d_in[21];
  const float* ffn_w1 = (const float*)d_in[22];
  const float* ffn_w2 = (const float*)d_in[23];
  const float* ln2_g  = (const float*)d_in[24];
  const float* ln2_b  = (const float*)d_in[25];

  float* ws = (float*)d_ws;
  float* Qp = ws + OFF_QP;
  float* Kp = ws + OFF_KP;
  float* Vp = ws + OFF_VP;
  float* S  = ws + OFF_S;
  unsigned* gmax = (unsigned*)(ws + OFF_GMAX);
  float* pattn = ws + OFF_PATTN;
  float* hc = ws + OFF_HC;
  float* KV = ws + OFF_KV;
  float* qbuf = ws + OFF_QBUF;

  init_gmax_kernel<<<1, 64, 0, stream>>>(gmax);
  // p-attention projections (2048x64x64 each), K scaled by DH^-0.5
  gemm_k64_bf16<<<dim3(64, 1), 256, 0, stream>>>(p, wq_p, Qp, 64, 0, nullptr, nullptr);
  gemm_k64_bf16<<<dim3(64, 1), 256, 0, stream>>>(p, wk_p, Kp, 64, 1, nullptr, nullptr);
  gemm_k64_bf16<<<dim3(64, 1), 256, 0, stream>>>(p, wv_p, Vp, 64, 0, nullptr, nullptr);
  // edge-modulated scores: single async-DMA pass over e (134MB)
  edge_scores_kernel<<<B_ * N_, 256, 0, stream>>>(e, we_p, Qp, Kp, S, gmax);
  // exp/mask/k_RW weighting + attention-apply (WMMA with ones-column denom)
  p_attn_apply_kernel<<<B_ * H_ * 16, 32, 0, stream>>>(S, gmax, k_RW, mask, Vp, pattn);
  // h_c = h + p + tanh(p_attn @ wo_p)
  gemm_k64_bf16<<<dim3(64, 1), 256, 0, stream>>>(pattn, wo_p, hc, 64, 2, hfeat, p);
  // cross-attention K/V projection (2048x64x128)
  gemm_k64_bf16<<<dim3(64, 2), 256, 0, stream>>>(hc, cq_wkv, KV, 128, 0, nullptr, nullptr);
  // perceiver cross-attn + residual
  cross_attn_kernel<<<B_, 256, 0, stream>>>(queries, cq_wq, KV, cq_wo, cq_bo, qbuf);
  // latent self-attn x2, O_queries, LN, FFN, LN
  latent_block_kernel<<<B_, 256, 0, stream>>>(qbuf, sa_wq, sa_wkv, sa_wo, sa_bo,
                                              oq_w, ln1_g, ln1_b, ffn_w1, ffn_w2,
                                              ln2_g, ln2_b, (float*)d_out);
}
